// CudaW4A16Linear_38783554683070
// MI455X (gfx1250) — matrix operand order repaired and checked by reading, not compiled, not run on hardware
//
#include <hip/hip_runtime.h>

typedef __attribute__((ext_vector_type(16))) _Float16 v16h;
typedef __attribute__((ext_vector_type(8)))  _Float16 v8h;
typedef __attribute__((ext_vector_type(2)))  _Float16 v2h;
typedef __attribute__((ext_vector_type(8)))  float    v8f;

// Problem constants (from reference setup_inputs)
constexpr int K_DIM  = 4096;
constexpr int O_DIM  = 11008;
constexpr int GROUP  = 128;             // quant group along K
constexpr int G_CNT  = K_DIM / GROUP;   // 32 groups
constexpr int GP_CNT = (G_CNT + 7) / 8; // 4 packed-zero words per O row
constexpr int KWORDS = K_DIM / 8;       // 512 packed weight words per O row

constexpr int TILES_N = O_DIM / 16;     // 688
constexpr int WAVES_PER_BLOCK = 8;      // 256 threads, wave32
constexpr int NBLK_N = TILES_N / WAVES_PER_BLOCK;   // 86
constexpr int KSPLIT = 2;               // two K-halves -> 2 fp32 partials/output
constexpr int GRP_PER_SPLIT = G_CNT / KSPLIT;       // 16 groups per wave
constexpr int NCHUNK = (K_DIM / KSPLIT) / 32;       // 64 K-chunks per wave

// v_perm_b32 selectors: dst = [ev.b[i], 0x00, od.b[i], 0x00]  (code 0xC -> 0x00)
__constant__ const unsigned kPermSel[4] = {
    0x0C040C00u, 0x0C050C01u, 0x0C060C02u, 0x0C070C03u };

__global__ __launch_bounds__(256)
void zero_out_kernel(float4* __restrict__ p, int n4) {
    int i = blockIdx.x * blockDim.x + threadIdx.x;
    if (i < n4) p[i] = make_float4(0.f, 0.f, 0.f, 0.f);
}

__global__ __launch_bounds__(256, 1)
void w4a16_wmma_kernel(const _Float16* __restrict__ x,
                       const int*      __restrict__ qweight,
                       const int*      __restrict__ qzeros,
                       const _Float16* __restrict__ scales,
                       float*          __restrict__ out)
{
    const int lane  = threadIdx.x & 31;
    const int wave  = threadIdx.x >> 5;
    const int l16   = lane & 15;
    const int lhalf = lane >> 4;                 // 0: lanes 0-15, 1: lanes 16-31

    const int kh     = blockIdx.x / NBLK_N;      // 0..1  (K-half)
    const int tile_n = (blockIdx.x % NBLK_N) * WAVES_PER_BLOCK + wave;

    const int n = tile_n * 16 + l16;             // B column (output feature)

    const _Float16*     xb   = x + (size_t)l16 * K_DIM;   // A row l16 of tile 0
    const unsigned int* wrow = (const unsigned int*)(qweight + (size_t)n * KWORDS);
    const unsigned int* zrow = (const unsigned int*)(qzeros  + (size_t)n * GP_CNT);
    const _Float16*     srow = scales + (size_t)n * G_CNT;

    const int cbase = kh * NCHUNK;               // first global chunk index
    const int gbase = kh * GRP_PER_SPLIT;        // first quant group

    v8f acc[4] = {};   // four 16x16 M-tiles -> all of M=64 shares one B dequant

    // ---- pipeline prologue: loads for chunk 0 ----
    v8h alo[4], ahi[4];
    unsigned long long qq;
    {
        const int kc = cbase * 32;
        qq = *(const unsigned long long*)(wrow + (kc >> 3) + lhalf * 2);
#pragma unroll
        for (int t = 0; t < 4; ++t) {
            const _Float16* xr = xb + (size_t)(t * 16) * K_DIM;
            alo[t] = *(const v8h*)(xr + kc + lhalf * 16);
            ahi[t] = *(const v8h*)(xr + kc + lhalf * 16 + 8);
        }
    }

    v2h s2 = {}, zo2 = {};

#pragma unroll 4
    for (int ic = 0; ic < NCHUNK; ++ic) {
        if ((ic & 3) == 0) {                     // static under unroll-4
            // Per-group dequant constants (shared by 4 chunks).
            const int g = gbase + (ic >> 2);
            const unsigned int zword = zrow[g >> 3];
            const unsigned int zn = (zword >> ((g & 7) * 4)) & 0xF;
            const _Float16 s  = srow[g];
            const _Float16 zo = (_Float16)(float)(1024u + zn);  // exact in fp16
            s2[0] = s;   s2[1] = s;
            zo2[0] = zo; zo2[1] = zo;
            // Prefetch next group's packed weights (global_prefetch_b8).
            if (g + 1 < gbase + GRP_PER_SPLIT)
                __builtin_prefetch(wrow + (((g + 1) * GROUP) >> 3), 0, 0);
        }

        // ---- stage next chunk's loads (wrap on last iter: dead values) ----
        const int kn = (cbase + ((ic + 1) & (NCHUNK - 1))) * 32;
        unsigned long long nqq =
            *(const unsigned long long*)(wrow + (kn >> 3) + lhalf * 2);
        v8h nlo[4], nhi[4];
#pragma unroll
        for (int t = 0; t < 4; ++t) {
            const _Float16* xr = xb + (size_t)(t * 16) * K_DIM;
            nlo[t] = *(const v8h*)(xr + kn + lhalf * 16);
            nhi[t] = *(const v8h*)(xr + kn + lhalf * 16 + 8);
        }

        // ---- B fragment: 16 int4 -> fp16 via 0x6400 magic + v_perm ----
        union { unsigned w[8]; v16h v; } bb;
#pragma unroll
        for (int j = 0; j < 2; ++j) {
            const unsigned q  = (unsigned)(qq >> (32 * j));
            const unsigned ev = q & 0x0F0F0F0Fu;          // nibbles 0,2,4,6
            const unsigned od = (q >> 4) & 0x0F0F0F0Fu;   // nibbles 1,3,5,7
#pragma unroll
            for (int i = 0; i < 4; ++i) {
                // p = fp16x2{1024+q[2i], 1024+q[2i+1]}
                unsigned p = __builtin_amdgcn_perm(od, ev, kPermSel[i])
                           | 0x64006400u;
                v2h h = __builtin_bit_cast(v2h, p);
                h = (h - zo2) * s2;              // exact sub, then v_pk_mul_f16
                bb.w[j * 4 + i] = __builtin_bit_cast(unsigned, h);
            }
        }

        // ---- 4 WMMAs: all M-tiles reuse the dequantized B fragment ----
        // A layout (16x32 f16): lanes 0-15 elems 0-15 -> K+0..15
        //                       lanes16-31 elems 0-15 -> K+16..31
#pragma unroll
        for (int t = 0; t < 4; ++t) {
            v16h a;
#pragma unroll
            for (int e = 0; e < 8; ++e) { a[e] = alo[t][e]; a[e + 8] = ahi[t][e]; }
            acc[t] = __builtin_amdgcn_wmma_f32_16x16x32_f16(
                false, a, false, bb.v, (short)0, acc[t], false, false);
        }

        // ---- rotate pipeline buffers (renamed away by unrolling) ----
        qq = nqq;
#pragma unroll
        for (int t = 0; t < 4; ++t) { alo[t] = nlo[t]; ahi[t] = nhi[t]; }
    }

    // C/D layout: lanes 0-15 hold N=l16; VGPR r holds row M = r + 8*lhalf.
    // Two K-half partials per element, combined with native f32 atomics
    // (order-invariant: fp32 add of exactly two addends to 0.0 is commutative).
    const size_t col = (size_t)tile_n * 16 + l16;
#pragma unroll
    for (int t = 0; t < 4; ++t) {
#pragma unroll
        for (int r = 0; r < 8; ++r) {
            const int mr = t * 16 + r + lhalf * 8;
            unsafeAtomicAdd(&out[(size_t)mr * O_DIM + col], acc[t][r]);
        }
    }
}

extern "C" void kernel_launch(void* const* d_in, const int* in_sizes, int n_in,
                              void* d_out, int out_size, void* d_ws, size_t ws_size,
                              hipStream_t stream) {
    (void)in_sizes; (void)n_in; (void)d_ws; (void)ws_size;
    const _Float16* x       = (const _Float16*)d_in[0];
    const int*      qweight = (const int*)d_in[1];
    const int*      qzeros  = (const int*)d_in[2];
    const _Float16* scales  = (const _Float16*)d_in[3];
    float*          out     = (float*)d_out;

    // Zero the output (atomic accumulation target); runs first on same stream.
    const int n4 = out_size / 4;                 // 704512 / 4 = 176128
    zero_out_kernel<<<(n4 + 255) / 256, 256, 0, stream>>>((float4*)out, n4);

    dim3 grid(KSPLIT * NBLK_N);   // 2 * 86 = 172 blocks (1376 waves)
    dim3 block(256);              // 8 wave32 waves
    w4a16_wmma_kernel<<<grid, block, 0, stream>>>(x, qweight, qzeros, scales, out);
}
